// MultiQueryAttention_24541443129578
// MI455X (gfx1250) — compile-verified, matrix operand order checked by reading, not run on hardware
//
#include <hip/hip_runtime.h>
#include <hip/hip_bf16.h>

// ---------------------------------------------------------------------------
// Types for CDNA5 WMMA (wave32)
// ---------------------------------------------------------------------------
typedef __attribute__((ext_vector_type(16))) __bf16 v16bf;
typedef __attribute__((ext_vector_type(8)))  __bf16 v8bf;
typedef __attribute__((ext_vector_type(8)))  float  v8f;
typedef __attribute__((ext_vector_type(4)))  unsigned int u32x4;
typedef __attribute__((ext_vector_type(8)))  int i32x8;
typedef __attribute__((ext_vector_type(4)))  int i32x4;

#define KSTR 72   // LDS row stride (bf16) for 64-elem rows: 64 + 8 pad = 144B

#if defined(__HIP_DEVICE_COMPILE__) && \
    __has_builtin(__builtin_amdgcn_tensor_load_to_lds) && \
    __has_builtin(__builtin_amdgcn_s_wait_tensorcnt)
#define HAVE_TDM 1
#else
#define HAVE_TDM 0
#endif

#if defined(__HIP_DEVICE_COMPILE__) && __has_builtin(__builtin_amdgcn_update_dpp)
#define HAVE_DPP 1
#else
#define HAVE_DPP 0
#endif

__device__ __forceinline__ v8f wmma_bf16(v16bf a, v16bf b, v8f c) {
    return __builtin_amdgcn_wmma_f32_16x16x32_bf16(
        false, a, false, b, (short)0, c, false, false);
}

// 16x32 bf16 A/B fragment from a row-major bf16 row: lanes 0-15 hold
// K={0..7,16..23}, lanes 16-31 hold K={8..15,24..31} -> two 16B ds loads.
__device__ __forceinline__ v16bf load_frag(const __bf16* rowPtr, int half) {
    v8bf lo = *(const v8bf*)(rowPtr + half * 8);
    v8bf hi = *(const v8bf*)(rowPtr + 16 + half * 8);
    v16bf r;
#pragma unroll
    for (int i = 0; i < 8; ++i) { r[i] = lo[i]; r[i + 8] = hi[i]; }
    return r;
}

__device__ __forceinline__ unsigned int pack_bf16(float a, float b) {
    union { __bf16 h[2]; unsigned int u; } x;
    x.h[0] = (__bf16)a; x.h[1] = (__bf16)b;
    return x.u;
}

// Max-reduction across the 16-lane row group (DPP butterfly, no LDS traffic).
__device__ __forceinline__ float red16_max(float v) {
#if HAVE_DPP
    int x = __float_as_int(v);
    v = fmaxf(v, __int_as_float(__builtin_amdgcn_update_dpp(x, x, 0x140, 0xF, 0xF, false))); // row_mirror
    x = __float_as_int(v);
    v = fmaxf(v, __int_as_float(__builtin_amdgcn_update_dpp(x, x, 0x141, 0xF, 0xF, false))); // row_half_mirror
    x = __float_as_int(v);
    v = fmaxf(v, __int_as_float(__builtin_amdgcn_update_dpp(x, x, 0x1B, 0xF, 0xF, false)));  // quad reverse
    x = __float_as_int(v);
    v = fmaxf(v, __int_as_float(__builtin_amdgcn_update_dpp(x, x, 0xB1, 0xF, 0xF, false)));  // pair swap
    return v;
#else
    v = fmaxf(v, __shfl_xor(v, 1, 32));
    v = fmaxf(v, __shfl_xor(v, 2, 32));
    v = fmaxf(v, __shfl_xor(v, 4, 32));
    v = fmaxf(v, __shfl_xor(v, 8, 32));
    return v;
#endif
}

// ---------------------------------------------------------------------------
// TDM: 2D bf16 tile (rows x rowElems) from global (row stride strideElems)
// into LDS at byte offset ldsOff; LDS rows padded per padIntervalEnc/padDwords.
// TDM ignores EXEC -> issue from one wave only.
// ---------------------------------------------------------------------------
__device__ __forceinline__ void tdm_load_tile_bf16(
    unsigned int ldsOff, const __bf16* gsrc, unsigned int rowElems,
    unsigned int rows, unsigned long long strideElems,
    int padIntervalEnc, int padDwords) {
#if HAVE_TDM
    unsigned long long ga = (unsigned long long)(size_t)gsrc;
    u32x4 g0;
    g0[0] = 1u;                                             // count=1, user mode
    g0[1] = ldsOff;                                         // lds_addr (bytes)
    g0[2] = (unsigned int)ga;                               // global_addr[31:0]
    g0[3] = (unsigned int)((ga >> 32) & 0x1FFFFFFu) | (2u << 30); // [56:32]|type=2
    const unsigned int padEn = (padIntervalEnc >= 0) ? 1u : 0u;
    const unsigned int pi = (padIntervalEnc >= 0) ? (unsigned)padIntervalEnc : 0u;
    const unsigned int pa = (padDwords > 0) ? (unsigned)(padDwords - 1) : 0u;
    i32x8 g1;
    g1[0] = (int)((1u << 16) | (padEn << 20) | (pi << 22) | (pa << 25)); // data=2B
    g1[1] = (int)((rowElems & 0xFFFFu) << 16);              // tensor_dim0 lo16
    g1[2] = (int)(((rowElems >> 16) & 0xFFFFu) | ((rows & 0xFFFFu) << 16));
    g1[3] = (int)(((rows >> 16) & 0xFFFFu) | ((rowElems & 0xFFFFu) << 16)); // tile_dim0
    g1[4] = (int)(rows & 0xFFFFu);                          // tile_dim1, tile_dim2=0
    g1[5] = (int)(unsigned int)(strideElems & 0xFFFFFFFFull);      // dim0_stride lo32
    g1[6] = (int)(unsigned int)((strideElems >> 32) & 0xFFFFull);  // dim0_stride hi16
    g1[7] = 0;
    i32x4 gz = {0, 0, 0, 0};
#if __clang_major__ >= 23
    i32x8 gz8 = {0, 0, 0, 0, 0, 0, 0, 0};
    __builtin_amdgcn_tensor_load_to_lds(g0, g1, gz, gz, gz8, 0);
#else
    __builtin_amdgcn_tensor_load_to_lds(g0, g1, gz, gz, 0);
#endif
#else
    (void)ldsOff; (void)gsrc; (void)rowElems; (void)rows; (void)strideElems;
    (void)padIntervalEnc; (void)padDwords;
#endif
}

__device__ __forceinline__ void tdm_wait() {
#if HAVE_TDM
    __builtin_amdgcn_s_wait_tensorcnt(0);
#endif
}

// Fallback cooperative bf16 tile copy (16B chunks), same padded layout.
__device__ __forceinline__ void copy_tile_bf16(
    __bf16* dst, int dstStride, const __bf16* src, size_t srcStride,
    int rows, int rowElems, int tid, int nthreads) {
    const int cpr = rowElems >> 3;
    const int total = rows * cpr;
    for (int idx = tid; idx < total; idx += nthreads) {
        int r = idx / cpr, c = idx - r * cpr;
        *(v8bf*)(dst + r * dstStride + c * 8) =
            *(const v8bf*)(src + (size_t)r * srcStride + c * 8);
    }
}

// ---------------------------------------------------------------------------
// Elementwise f32 -> bf16 (4 per thread, packed b64 stores)
// ---------------------------------------------------------------------------
__global__ __launch_bounds__(256) void cvt_bf16_kernel(
    const float* __restrict__ X, __bf16* __restrict__ Y) {
    size_t idx = ((size_t)blockIdx.x * 256 + threadIdx.x) * 4;
    float4 v = *(const float4*)(X + idx);
    uint2 u;
    u.x = pack_bf16(v.x, v.y);
    u.y = pack_bf16(v.z, v.w);
    *(uint2*)(Y + idx) = u;
}

// ---------------------------------------------------------------------------
// Weight convert + transpose (+ optional scale): Wt[n][k] = (bf16)(s*W[k][n])
// ---------------------------------------------------------------------------
__global__ __launch_bounds__(256) void cvt_transpose_kernel(
    const float* __restrict__ W, __bf16* __restrict__ Wt, int Kdim, int N,
    float scale) {
    int idx = blockIdx.x * 256 + threadIdx.x;
    int k = idx / N, n = idx - k * N;
    Wt[(size_t)n * Kdim + k] = (__bf16)(W[idx] * scale);
}

// ---------------------------------------------------------------------------
// GEMM: C[M,N] = A[M,K] * Bt[N,K]^T, all-bf16 operands, TDM double-buffered.
// BK=64. 8 waves as 4x2; wave tile 32 x (16*WCOLS); WG tile 128 x (32*WCOLS).
// ---------------------------------------------------------------------------
template <int WCOLS, bool OUT_BF16>
__global__ __launch_bounds__(256) void gemm_wmma_kernel(
    const __bf16* __restrict__ A, const __bf16* __restrict__ Bt,
    void* __restrict__ Cv, int M, int N, int Kdim) {
    constexpr int BN = 32 * WCOLS;
    __shared__ __bf16 sA[2][128 * KSTR];
    __shared__ __bf16 sB[2][BN * KSTR];

    const int tid  = threadIdx.x;
    const int lane = tid & 31;
    const int w    = tid >> 5;
    const int l15  = lane & 15;
    const int half = lane >> 4;
    const int m0   = blockIdx.y * 128;
    const int n0   = blockIdx.x * BN;
    const int sm   = (w >> 1) * 32;
    const int sn   = (w & 1) * (16 * WCOLS);

    auto stage = [&](int buf, int k0) {
#if HAVE_TDM
        if (w == 0) {
            tdm_load_tile_bf16((unsigned int)(size_t)(void*)&sA[buf][0],
                               A + (size_t)m0 * Kdim + k0, 64, 128,
                               (unsigned long long)Kdim, 4, 4);
            tdm_load_tile_bf16((unsigned int)(size_t)(void*)&sB[buf][0],
                               Bt + (size_t)n0 * Kdim + k0, 64, BN,
                               (unsigned long long)Kdim, 4, 4);
        }
#else
        copy_tile_bf16(&sA[buf][0], KSTR, A + (size_t)m0 * Kdim + k0,
                       (size_t)Kdim, 128, 64, tid, 256);
        copy_tile_bf16(&sB[buf][0], KSTR, Bt + (size_t)n0 * Kdim + k0,
                       (size_t)Kdim, BN, 64, tid, 256);
#endif
    };

    v8f acc[2][WCOLS];
#pragma unroll
    for (int t = 0; t < 2; ++t)
#pragma unroll
        for (int u = 0; u < WCOLS; ++u)
#pragma unroll
            for (int r = 0; r < 8; ++r) acc[t][u][r] = 0.0f;

    stage(0, 0);
    if (w == 0) tdm_wait();
    __syncthreads();

    int cur = 0;
    for (int k0 = 0; k0 < Kdim; k0 += 64) {
        if (k0 + 64 < Kdim) stage(cur ^ 1, k0 + 64);   // overlap DMA with WMMA

#pragma unroll
        for (int kk = 0; kk < 2; ++kk) {
            v16bf aF[2], bF[WCOLS];
#pragma unroll
            for (int t = 0; t < 2; ++t)
                aF[t] = load_frag(&sA[cur][(sm + t * 16 + l15) * KSTR + kk * 32], half);
#pragma unroll
            for (int u = 0; u < WCOLS; ++u)
                bF[u] = load_frag(&sB[cur][(sn + u * 16 + l15) * KSTR + kk * 32], half);
#pragma unroll
            for (int t = 0; t < 2; ++t)
#pragma unroll
                for (int u = 0; u < WCOLS; ++u)
                    acc[t][u] = wmma_bf16(aF[t], bF[u], acc[t][u]);
        }
        if (w == 0) tdm_wait();
        __syncthreads();
        cur ^= 1;
    }

#pragma unroll
    for (int t = 0; t < 2; ++t)
#pragma unroll
        for (int u = 0; u < WCOLS; ++u)
#pragma unroll
            for (int r = 0; r < 8; ++r) {
                int m = m0 + sm + t * 16 + r + 8 * half;
                int n = n0 + sn + u * 16 + l15;
                float v = acc[t][u][r];
                if (OUT_BF16) ((__bf16*)Cv)[(size_t)m * N + n] = (__bf16)v;
                else          ((float*)Cv)[(size_t)m * N + n] = v;
            }
}

// ---------------------------------------------------------------------------
// Causal MQA flash attention, 64-key blocks, double-buffered K (TDM) and
// transposed-V tiles. q: [B,S,H*64] bf16 (pre-scaled by 1/sqrt(dk) via W_q);
// k,v: [B,S,64] bf16 shared head. Wave owns 16 q rows; 8 waves/WG = 128-row
// q block per (b,head). Row sums via WMMA against an all-ones fragment.
// ---------------------------------------------------------------------------
__global__ __launch_bounds__(256) void flash_mqa_kernel(
    const __bf16* __restrict__ qb, const __bf16* __restrict__ kb,
    const __bf16* __restrict__ vb, __bf16* __restrict__ ob, int S) {
    __shared__ __bf16 sK[2][64 * KSTR];    // K tile [j][d], padded rows
    __shared__ __bf16 svt[2][64 * KSTR];   // V tile transposed [d][j]
    __shared__ __bf16 sP[8 * 16 * 64];     // per-wave P tile [16][64]

    const int tid  = threadIdx.x;
    const int lane = tid & 31;
    const int w    = tid >> 5;
    const int l15  = lane & 15;
    const int half = lane >> 4;
    const int q0   = blockIdx.x * 128;
    const int head = blockIdx.y;
    const int b    = blockIdx.z;
    const int i0   = q0 + w * 16;

    const size_t rowbase = (size_t)b * S * 1024;
    const size_t kvbase  = (size_t)b * S * 64;

    auto stageK = [&](int buf, int kt) {
#if HAVE_TDM
        if (w == 0)
            tdm_load_tile_bf16((unsigned int)(size_t)(void*)&sK[buf][0],
                               kb + kvbase + (size_t)kt * 64, 64, 64, 64ull, 4, 4);
#else
        copy_tile_bf16(&sK[buf][0], KSTR, kb + kvbase + (size_t)kt * 64, 64ull,
                       64, 64, tid, 256);
#endif
    };
    auto stageV = [&](int buf, int kt) {
#pragma unroll
        for (int it = 0; it < 8; ++it) {
            int idx = tid + it * 256;          // 64 d x 32 j-pairs / 256 = 8
            int d = idx & 63, jp = idx >> 6;
            float a = (float)vb[kvbase + (size_t)(kt + 2 * jp) * 64 + d];
            float c = (float)vb[kvbase + (size_t)(kt + 2 * jp + 1) * 64 + d];
            *(unsigned int*)&svt[buf][d * KSTR + 2 * jp] = pack_bf16(a, c);
        }
    };

    v16bf aQ[2];
#pragma unroll
    for (int dd = 0; dd < 2; ++dd)
        aQ[dd] = load_frag(qb + rowbase + (size_t)(i0 + l15) * 1024 + head * 64 + dd * 32, half);

    v16bf onesF;
#pragma unroll
    for (int i = 0; i < 16; ++i) onesF[i] = (__bf16)1.0f;

    v8f O[4], Osum;
#pragma unroll
    for (int dc = 0; dc < 4; ++dc)
#pragma unroll
        for (int r = 0; r < 8; ++r) O[dc][r] = 0.0f;
#pragma unroll
    for (int r = 0; r < 8; ++r) Osum[r] = 0.0f;
    float mrow[8];
#pragma unroll
    for (int r = 0; r < 8; ++r) mrow[r] = -3.0e38f;

    __bf16* myP = &sP[w * 16 * 64];
    const int kend = q0 + 128;

    stageK(0, 0);
    stageV(0, 0);
    if (w == 0) tdm_wait();
    __syncthreads();

    int cur = 0;
    for (int kt = 0; kt < kend; kt += 64) {
        if (kt + 64 < kend) {                  // overlap next-tile staging
            stageK(cur ^ 1, kt + 64);
            stageV(cur ^ 1, kt + 64);
        }

        if (kt <= i0 + 15) {
            // ---- scores: four 16x16 j-tiles over 64 keys ----
            v8f sc[4];
#pragma unroll
            for (int jt = 0; jt < 4; ++jt) {
                const int jbase = kt + jt * 16;
                if (jbase <= i0 + 15) {          // wave-uniform: live columns
                    v8f s;
#pragma unroll
                    for (int r = 0; r < 8; ++r) s[r] = 0.0f;
#pragma unroll
                    for (int dd = 0; dd < 2; ++dd) {
                        v16bf bK = load_frag(&sK[cur][(jt * 16 + l15) * KSTR + dd * 32], half);
                        s = wmma_bf16(aQ[dd], bK, s);
                    }
                    if (jbase + 15 > i0) {       // diagonal tile: element mask
                        const int j = jbase + l15;
#pragma unroll
                        for (int r = 0; r < 8; ++r)
                            if (j > i0 + r + 8 * half) s[r] = -1.0e9f;
                    }
                    sc[jt] = s;
                } else {
#pragma unroll
                    for (int r = 0; r < 8; ++r) sc[jt][r] = -1.0e9f;
                }
            }

            // ---- online softmax: row max via DPP butterfly ----
            float mnew[8], cfac[8];
#pragma unroll
            for (int r = 0; r < 8; ++r) {
                float v = fmaxf(fmaxf(sc[0][r], sc[1][r]), fmaxf(sc[2][r], sc[3][r]));
                v = red16_max(v);
                mnew[r] = fmaxf(mrow[r], v);
                cfac[r] = __expf(mrow[r] - mnew[r]);
                mrow[r] = mnew[r];
            }
#pragma unroll
            for (int jt = 0; jt < 4; ++jt)
#pragma unroll
                for (int r = 0; r < 8; ++r) {
                    float p = __expf(sc[jt][r] - mnew[r]);
                    myP[(r + 8 * half) * 64 + jt * 16 + l15] = (__bf16)p;
                }
#pragma unroll
            for (int dc = 0; dc < 4; ++dc)
#pragma unroll
                for (int r = 0; r < 8; ++r) O[dc][r] *= cfac[r];
#pragma unroll
            for (int r = 0; r < 8; ++r) Osum[r] *= cfac[r];

            asm volatile("" ::: "memory");     // keep ds stores before reloads
            v16bf pF[2];
#pragma unroll
            for (int jj = 0; jj < 2; ++jj)
                pF[jj] = load_frag(myP + l15 * 64 + jj * 32, half);
#pragma unroll
            for (int jj = 0; jj < 2; ++jj) {
#pragma unroll
                for (int dc = 0; dc < 4; ++dc) {
                    v16bf vF = load_frag(&svt[cur][(dc * 16 + l15) * KSTR + jj * 32], half);
                    O[dc] = wmma_bf16(pF[jj], vF, O[dc]);
                }
                Osum = wmma_bf16(pF[jj], onesF, Osum);
            }
        }
        if (w == 0) tdm_wait();
        __syncthreads();
        cur ^= 1;
    }

#pragma unroll
    for (int dc = 0; dc < 4; ++dc)
#pragma unroll
        for (int r = 0; r < 8; ++r) {
            const int i = i0 + r + 8 * half;
            float v = O[dc][r] / Osum[r];
            ob[rowbase + (size_t)i * 1024 + head * 64 + dc * 16 + l15] = (__bf16)v;
        }
}

// ---------------------------------------------------------------------------
// Launcher
// ---------------------------------------------------------------------------
extern "C" void kernel_launch(void* const* d_in, const int* in_sizes, int n_in,
                              void* d_out, int out_size, void* d_ws, size_t ws_size,
                              hipStream_t stream) {
    (void)n_in; (void)out_size; (void)ws_size;
    const float* Q   = (const float*)d_in[0];
    const float* Kin = (const float*)d_in[1];
    const float* Vin = (const float*)d_in[2];
    const float* W_q = (const float*)d_in[3];
    const float* W_k = (const float*)d_in[4];
    const float* W_v = (const float*)d_in[5];
    const float* W_o = (const float*)d_in[6];
    float* out = (float*)d_out;

    const int DM = 1024;
    const int S  = 2048;
    const int BS = in_sizes[0] / DM;     // B*S = 4096
    const int B  = BS / S;
    const int H  = 16;

    __bf16* qb  = (__bf16*)d_ws;                       // [BS,1024] projected q
    __bf16* kb  = qb  + (size_t)BS * 1024;             // [BS,64]
    __bf16* vb  = kb  + (size_t)BS * 64;               // [BS,64]
    __bf16* Qc  = vb  + (size_t)BS * 64;               // [BS,1024] bf16 Q; reused as ob
    __bf16* ob  = Qc;                                  // alias: Qc dead before flash
    __bf16* Kc  = Qc  + (size_t)BS * 1024;             // [BS,1024] bf16 K input
    __bf16* Vc  = Kc  + (size_t)BS * 1024;             // [BS,1024] bf16 V input
    __bf16* wqT = Vc  + (size_t)BS * 1024;             // [1024,1024] (N,K)
    __bf16* wkT = wqT + (size_t)DM * DM;               // [64,1024]
    __bf16* wvT = wkT + (size_t)64 * DM;               // [64,1024]
    __bf16* woT = wvT + (size_t)64 * DM;               // [1024,1024]

    dim3 blk(256);
    const int cvtBlocks = (BS * DM) / (256 * 4);
    // bf16 copies of activations (enables all-TDM GEMM staging)
    cvt_bf16_kernel<<<dim3(cvtBlocks), blk, 0, stream>>>(Q, Qc);
    cvt_bf16_kernel<<<dim3(cvtBlocks), blk, 0, stream>>>(Kin, Kc);
    cvt_bf16_kernel<<<dim3(cvtBlocks), blk, 0, stream>>>(Vin, Vc);
    // Weight convert+transpose to bf16 [N][K]; fold 1/sqrt(dk) into W_q
    cvt_transpose_kernel<<<dim3((DM * DM) / 256), blk, 0, stream>>>(W_q, wqT, DM, DM, 0.125f);
    cvt_transpose_kernel<<<dim3((DM * 64) / 256), blk, 0, stream>>>(W_k, wkT, DM, 64, 1.0f);
    cvt_transpose_kernel<<<dim3((DM * 64) / 256), blk, 0, stream>>>(W_v, wvT, DM, 64, 1.0f);
    cvt_transpose_kernel<<<dim3((DM * DM) / 256), blk, 0, stream>>>(W_o, woT, DM, DM, 1.0f);

    // Projections (all-bf16, TDM double-buffered)
    gemm_wmma_kernel<4, true><<<dim3(DM / 128, BS / 128), blk, 0, stream>>>(
        Qc, wqT, qb, BS, DM, DM);
    gemm_wmma_kernel<2, true><<<dim3(1, BS / 128), blk, 0, stream>>>(
        Kc, wkT, kb, BS, 64, DM);
    gemm_wmma_kernel<2, true><<<dim3(1, BS / 128), blk, 0, stream>>>(
        Vc, wvT, vb, BS, 64, DM);
    // Causal MQA flash attention
    flash_mqa_kernel<<<dim3(S / 128, H, B), blk, 0, stream>>>(qb, kb, vb, ob, S);
    // Output projection (f32 result)
    gemm_wmma_kernel<4, false><<<dim3(DM / 128, BS / 128), blk, 0, stream>>>(
        ob, woT, out, BS, DM, DM);
}
